// MambaPool_76888504533065
// MI455X (gfx1250) — compile-verified, matrix operand order checked by reading, not run on hardware
//
#include <hip/hip_runtime.h>

// ---------------------------------------------------------------------------
// Mamba2 graph-pool pipeline for gfx1250 (MI455X).
// bf16 WMMA GEMMs with LDS-staged (async-copy double-buffered) A operand.
// ---------------------------------------------------------------------------

#define NGRAPHS   16
#define MAX_L     320
#define D_MODEL   256
#define D_STATE   32
#define D_CONV    4
#define D_INNER   1024
#define HEADDIM   64
#define NHEADS    16
#define CONV_DIM  1088          // D_INNER + 2*D_STATE
#define D_IN_PROJ 2128          // 2*D_INNER + 2*D_STATE + NHEADS
#define ZX_LD     2176          // D_IN_PROJ padded to a multiple of 128
#define M_TOT     (NGRAPHS * MAX_L)   // 5120 padded rows
#define EPSF      1e-5f

typedef __attribute__((ext_vector_type(16))) __bf16 v16bf;
typedef __attribute__((ext_vector_type(8)))  float  v8f;

// 128-bit vector type matching the async-LDS builtin's pointee type
typedef int v4i __attribute__((vector_size(16)));
typedef __attribute__((address_space(1))) v4i* as1_v4i;
typedef __attribute__((address_space(3))) v4i* as3_v4i;

union Frag16 { v16bf v; uint4 q[2]; };

#if __has_builtin(__builtin_amdgcn_global_load_async_to_lds_b128)
#define HAS_ASYNC_LDS 1
#endif

#if __has_builtin(__builtin_amdgcn_s_wait_asynccnt)
#define WAIT_ASYNC(n) __builtin_amdgcn_s_wait_asynccnt(n)
#else
#define WAIT_ASYNC(n) asm volatile("s_wait_asynccnt %0" ::"i"(n) : "memory")
#endif

__device__ __forceinline__ void stage_a16(const unsigned short* __restrict__ src,
                                          uint4* dst) {
#ifdef HAS_ASYNC_LDS
  __builtin_amdgcn_global_load_async_to_lds_b128(
      (as1_v4i)(unsigned long long)(const void*)src,
      (as3_v4i)(unsigned int)(unsigned long long)(void*)dst,
      0, 0);
#else
  *dst = *(const uint4*)src;
#endif
}

__device__ __forceinline__ unsigned short f32_to_bf16(float f) {
  unsigned int u = __float_as_uint(f);
  u += 0x7fffu + ((u >> 16) & 1u);           // round-to-nearest-even
  return (unsigned short)(u >> 16);
}

__device__ __forceinline__ float siluf(float x) {
  return x / (1.f + __expf(-x));
}

// ---------------------------------------------------------------------------
// Small utility kernels
// ---------------------------------------------------------------------------
__global__ void zero_i32(int* p, int n) {
  int i = blockIdx.x * blockDim.x + threadIdx.x;
  if (i < n) p[i] = 0;
}

__global__ void zero_u16(unsigned short* p, int n) {
  int i = blockIdx.x * blockDim.x + threadIdx.x;
  if (i < n) p[i] = 0;
}

__global__ void count_lens(const int* __restrict__ batch, int* lens, int n) {
  int i = blockIdx.x * blockDim.x + threadIdx.x;
  if (i < n) atomicAdd(&lens[batch[i]], 1);
}

__global__ void compute_starts(const int* __restrict__ lens, int* starts) {
  if (threadIdx.x == 0) {
    int s = 0;
    for (int g = 0; g < NGRAPHS; ++g) { starts[g] = s; s += lens[g]; }
  }
}

// scatter node features into padded bf16 [NGRAPHS*MAX_L, D_MODEL]
__global__ void scatter_x(const float* __restrict__ x,
                          const int* __restrict__ batch,
                          const int* __restrict__ starts,
                          unsigned short* __restrict__ padded) {
  int i = blockIdx.x;           // node index
  int c = threadIdx.x;          // channel 0..255
  int g = batch[i];
  int pos = i - starts[g];
  padded[(size_t)(g * MAX_L + pos) * D_MODEL + c] = f32_to_bf16(x[(size_t)i * D_MODEL + c]);
}

__global__ void convert_f32_bf16(const float* __restrict__ in,
                                 unsigned short* __restrict__ out, int n) {
  int i = blockIdx.x * blockDim.x + threadIdx.x;
  if (i < n) out[i] = f32_to_bf16(in[i]);
}

// ---------------------------------------------------------------------------
// bf16 WMMA GEMM:  C[M,N] (f32) = A[M,K] (bf16, row major) * W[N,K]^T (bf16)
//
// Block = 256 threads (8 waves) -> 64 (M) x 128 (N) C tile.
// The shared 64xK A stripe is double-buffered through LDS in K=32 chunks
// (one async b128 per thread per chunk); each wave owns a 16-column slice,
// reusing the LDS A fragments across its 4 M-tiles and keeping its B
// fragment in registers.  Requires M%64==0, N%128==0, K%32==0.
// ---------------------------------------------------------------------------
__global__ __launch_bounds__(256)
void gemm_bf16_wmma(const unsigned short* __restrict__ A,
                    const unsigned short* __restrict__ W,
                    float* __restrict__ C,
                    int M, int N, int K) {
  __shared__ uint4 sA[512];                    // 2 buffers x 4KB (64 rows x 64B)

  const int tid  = threadIdx.x;
  const int wave = tid >> 5;
  const int lane = tid & 31;
  const int l16   = lane & 15;
  const int half8 = (lane >> 4) << 3;          // K sub-base within chunk: 0 or 8

  const int m0 = blockIdx.y << 6;              // 64-row stripe
  const int n0 = (blockIdx.x << 7) + (wave << 4);

  // per-thread A staging address: row = tid/4, 16B segment = tid%4
  const unsigned short* asrc =
      A + (size_t)(m0 + (tid >> 2)) * K + ((tid & 3) << 3);
  const unsigned short* wrow = W + (size_t)(n0 + l16) * K;

  v8f acc[4];
#pragma unroll
  for (int q = 0; q < 4; ++q) acc[q] = (v8f){0.f,0.f,0.f,0.f,0.f,0.f,0.f,0.f};

  const int KC = K >> 5;
  stage_a16(asrc, &sA[tid]);                   // prologue: chunk 0 -> buf 0

  for (int kc = 0; kc < KC; ++kc) {
    const int cur = kc & 1;
    if (kc + 1 < KC)
      stage_a16(asrc + ((kc + 1) << 5), &sA[((cur ^ 1) << 8) + tid]);
#ifdef HAS_ASYNC_LDS
    if (kc + 1 < KC) WAIT_ASYNC(1); else WAIT_ASYNC(0);
#endif
    __syncthreads();

    // B fragment for this chunk (registers, one per wave-column)
    const unsigned short* wptr = wrow + (kc << 5) + half8;
    Frag16 fb;
    fb.q[0] = *(const uint4*)(wptr);
    fb.q[1] = *(const uint4*)(wptr + 16);
    __builtin_prefetch(wptr + 32, 0, 3);       // next chunk of this B row

    const char* abuf = (const char*)&sA[cur << 8];
#pragma unroll
    for (int q = 0; q < 4; ++q) {
      const char* lptr = abuf + (((q << 4) + l16) << 6) + (half8 << 1);
      Frag16 fa;
      fa.q[0] = *(const uint4*)(lptr);
      fa.q[1] = *(const uint4*)(lptr + 32);
      acc[q] = __builtin_amdgcn_wmma_f32_16x16x32_bf16(
          false, fa.v, false, fb.v, (short)0, acc[q], false, false);
    }
    __syncthreads();
  }

#pragma unroll
  for (int q = 0; q < 4; ++q) {
    int mbase = m0 + (q << 4) + ((lane >> 4) << 3);
#pragma unroll
    for (int r = 0; r < 8; ++r)
      C[(size_t)(mbase + r) * N + n0 + l16] = acc[q][r];
  }
}

// ---------------------------------------------------------------------------
// Depthwise causal conv(4) + bias + SiLU on the xBC slice of zxbcdt.
// ---------------------------------------------------------------------------
__global__ void conv_silu(const float* __restrict__ zx,
                          const float* __restrict__ convw,
                          const float* __restrict__ convb,
                          float* __restrict__ xBC) {
  int idx = blockIdx.x * blockDim.x + threadIdx.x;   // M_TOT * CONV_DIM
  if (idx >= M_TOT * CONV_DIM) return;
  int c   = idx % CONV_DIM;
  int row = idx / CONV_DIM;
  int t   = row % MAX_L;
  const float* wc = convw + c * D_CONV;
  float acc = convb[c];
#pragma unroll
  for (int j = 0; j < D_CONV; ++j) {
    int tt = t - (D_CONV - 1) + j;
    if (tt >= 0)
      acc += wc[j] * zx[(size_t)(row - (D_CONV - 1) + j) * ZX_LD + D_INNER + c];
  }
  xBC[idx] = siluf(acc);
}

// dt = softplus(raw + bias); dA = exp(dt * -exp(A_log))
__global__ void dt_da_kernel(const float* __restrict__ zx,
                             const float* __restrict__ dt_bias,
                             const float* __restrict__ A_log,
                             float* __restrict__ dt,
                             float* __restrict__ dA) {
  int idx = blockIdx.x * blockDim.x + threadIdx.x;   // M_TOT * NHEADS
  if (idx >= M_TOT * NHEADS) return;
  int h   = idx & (NHEADS - 1);
  int row = idx >> 4;
  float xr = zx[(size_t)row * ZX_LD + (D_IN_PROJ - NHEADS) + h] + dt_bias[h];
  float sp = (xr > 20.f) ? xr : log1pf(__expf(xr));
  float a  = -__expf(A_log[h]);
  dt[idx] = sp;
  dA[idx] = __expf(sp * a);
}

// ---------------------------------------------------------------------------
// Selective scan: block = 64 threads = one (graph, head); thread p keeps the
// 32-wide state row h[p, 0:32] in registers; B/C staged via LDS each step.
// ---------------------------------------------------------------------------
__global__ __launch_bounds__(64)
void scan_kernel(const float* __restrict__ xBC,
                 const float* __restrict__ dt,
                 const float* __restrict__ dA,
                 const float* __restrict__ Dskip,
                 float* __restrict__ ybuf) {
  const int gh = blockIdx.x;
  const int g = gh >> 4, h = gh & 15;
  const int p = threadIdx.x;              // 0..63
  const float dsk = Dskip[h];

  float s[D_STATE];
#pragma unroll
  for (int n = 0; n < D_STATE; ++n) s[n] = 0.f;

  __shared__ float sB[D_STATE], sC[D_STATE];
  const int rowbase = g * MAX_L;

  for (int t = 0; t < MAX_L; ++t) {
    const int row = rowbase + t;
    __syncthreads();
    if (p < D_STATE) {
      sB[p] = xBC[(size_t)row * CONV_DIM + D_INNER + p];
      sC[p] = xBC[(size_t)row * CONV_DIM + D_INNER + D_STATE + p];
    }
    __syncthreads();
    const float dA_t = dA[row * NHEADS + h];
    const float dt_t = dt[row * NHEADS + h];
    const float xp   = xBC[(size_t)row * CONV_DIM + h * HEADDIM + p];
    const float dtx  = dt_t * xp;
    float y = 0.f;
#pragma unroll
    for (int n = 0; n < D_STATE; ++n) {
      s[n] = fmaf(s[n], dA_t, dtx * sB[n]);
      y = fmaf(s[n], sC[n], y);
    }
    ybuf[(size_t)row * D_INNER + h * HEADDIM + p] = y + dsk * xp;
  }
}

// ---------------------------------------------------------------------------
// y = y * silu(z); RMS norm over D_INNER; emit bf16 activations for out_proj.
// ---------------------------------------------------------------------------
__global__ __launch_bounds__(256)
void gate_rms(const float* __restrict__ ybuf,
              const float* __restrict__ zx,
              const float* __restrict__ rms_w,
              unsigned short* __restrict__ yn) {
  const int row = blockIdx.x;
  const int tid = threadIdx.x;
  __shared__ float red[256];
  float vals[4];
  float ss = 0.f;
#pragma unroll
  for (int i = 0; i < 4; ++i) {
    int c = tid + (i << 8);
    float z = zx[(size_t)row * ZX_LD + c];
    float y = ybuf[(size_t)row * D_INNER + c] * siluf(z);
    vals[i] = y;
    ss += y * y;
  }
  red[tid] = ss;
  __syncthreads();
  for (int o = 128; o > 0; o >>= 1) {
    if (tid < o) red[tid] += red[tid + o];
    __syncthreads();
  }
  const float scale = rsqrtf(red[0] * (1.f / (float)D_INNER) + EPSF);
#pragma unroll
  for (int i = 0; i < 4; ++i) {
    int c = tid + (i << 8);
    yn[(size_t)row * D_INNER + c] = f32_to_bf16(vals[i] * scale * rms_w[c]);
  }
}

// ---------------------------------------------------------------------------
// Masked mean pool over valid timesteps + LayerNorm over 256 channels.
// ---------------------------------------------------------------------------
__global__ __launch_bounds__(256)
void pool_ln(const float* __restrict__ seq,
             const int* __restrict__ lens,
             const float* __restrict__ ln_w,
             const float* __restrict__ ln_b,
             float* __restrict__ out) {
  const int g = blockIdx.x;
  const int c = threadIdx.x;
  const int len = lens[g];
  float sum = 0.f;
  for (int t = 0; t < len; ++t)
    sum += seq[(size_t)(g * MAX_L + t) * D_MODEL + c];
  const float pooled = sum / (float)len;

  __shared__ float red[256];
  red[c] = pooled;
  __syncthreads();
  for (int o = 128; o > 0; o >>= 1) {
    if (c < o) red[c] += red[c + o];
    __syncthreads();
  }
  const float mu = red[0] * (1.f / (float)D_MODEL);
  __syncthreads();
  const float d = pooled - mu;
  red[c] = d * d;
  __syncthreads();
  for (int o = 128; o > 0; o >>= 1) {
    if (c < o) red[c] += red[c + o];
    __syncthreads();
  }
  const float var = red[0] * (1.f / (float)D_MODEL);
  out[(size_t)g * D_MODEL + c] = d * rsqrtf(var + EPSF) * ln_w[c] + ln_b[c];
}

// ---------------------------------------------------------------------------
// Host-side orchestration
// ---------------------------------------------------------------------------
extern "C" void kernel_launch(void* const* d_in, const int* in_sizes, int n_in,
                              void* d_out, int out_size, void* d_ws, size_t ws_size,
                              hipStream_t stream) {
  const int N = in_sizes[0] / D_MODEL;   // number of nodes

  const float* x         = (const float*)d_in[0];
  const int*   batch     = (const int*)  d_in[1];
  const float* proj_in_w = (const float*)d_in[2];
  const float* ln_w      = (const float*)d_in[3];
  const float* ln_b      = (const float*)d_in[4];

  // workspace carve-out (256B aligned)
  char* p = (char*)d_ws;
  auto carve = [&](size_t bytes) -> void* {
    void* r = (void*)p;
    p += (bytes + 255) & ~(size_t)255;
    return r;
  };
  int*            lens   = (int*)carve(NGRAPHS * sizeof(int));
  int*            starts = (int*)carve(NGRAPHS * sizeof(int));
  unsigned short* Abf    = (unsigned short*)carve((size_t)M_TOT * D_INNER * 2);   // A operand
  unsigned short* Wproj  = (unsigned short*)carve((size_t)D_MODEL * D_MODEL * 2);
  unsigned short* Win    = (unsigned short*)carve((size_t)ZX_LD * D_MODEL * 2);   // padded rows
  unsigned short* Wout   = (unsigned short*)carve((size_t)D_MODEL * D_INNER * 2);
  float*          zx     = (float*)carve((size_t)M_TOT * ZX_LD * 4);
  float*          xbc    = (float*)carve((size_t)M_TOT * CONV_DIM * 4);
  float*          dt     = (float*)carve((size_t)M_TOT * NHEADS * 4);
  float*          dA     = (float*)carve((size_t)M_TOT * NHEADS * 4);
  float*          ybuf   = (float*)carve((size_t)M_TOT * D_INNER * 4);
  float*          seq    = (float*)carve((size_t)M_TOT * D_MODEL * 4);

  auto gemm = [&](const unsigned short* A, const unsigned short* W, float* C,
                  int M, int Nn, int K) {
    dim3 grid(Nn >> 7, M >> 6);
    gemm_bf16_wmma<<<grid, 256, 0, stream>>>(A, W, C, M, Nn, K);
  };

  // ---- stage 0: lens/starts, padded bf16 scatter, weight packs -------------
  zero_i32<<<1, 32, 0, stream>>>(lens, NGRAPHS);
  count_lens<<<(N + 255) / 256, 256, 0, stream>>>(batch, lens, N);
  compute_starts<<<1, 32, 0, stream>>>(lens, starts);
  zero_u16<<<(M_TOT * D_MODEL) / 256, 256, 0, stream>>>(Abf, M_TOT * D_MODEL);
  // zero the padded weight rows once per call (rows 2128..2175 -> zero output cols)
  zero_u16<<<((ZX_LD - D_IN_PROJ) * D_MODEL) / 256, 256, 0, stream>>>(
      Win + (size_t)D_IN_PROJ * D_MODEL, (ZX_LD - D_IN_PROJ) * D_MODEL);
  scatter_x<<<N, D_MODEL, 0, stream>>>(x, batch, starts, Abf);
  convert_f32_bf16<<<(D_MODEL * D_MODEL) / 256, 256, 0, stream>>>(proj_in_w, Wproj,
                                                                  D_MODEL * D_MODEL);

  // ---- h = padded_x @ proj_in_w^T -----------------------------------------
  gemm(Abf, Wproj, seq, M_TOT, D_MODEL, D_MODEL);

  // ---- two mamba2 blocks ---------------------------------------------------
  for (int b = 0; b < 2; ++b) {
    const float* in_proj_w  = (const float*)d_in[5 + b * 8];
    const float* conv_w     = (const float*)d_in[6 + b * 8];
    const float* conv_b     = (const float*)d_in[7 + b * 8];
    const float* dt_bias    = (const float*)d_in[8 + b * 8];
    const float* A_log      = (const float*)d_in[9 + b * 8];
    const float* Dskip      = (const float*)d_in[10 + b * 8];
    const float* rms_w      = (const float*)d_in[11 + b * 8];
    const float* out_proj_w = (const float*)d_in[12 + b * 8];

    convert_f32_bf16<<<(D_IN_PROJ * D_MODEL) / 256, 256, 0, stream>>>(
        in_proj_w, Win, D_IN_PROJ * D_MODEL);
    convert_f32_bf16<<<(D_MODEL * D_INNER) / 256, 256, 0, stream>>>(
        out_proj_w, Wout, D_MODEL * D_INNER);
    convert_f32_bf16<<<(M_TOT * D_MODEL) / 256, 256, 0, stream>>>(
        seq, Abf, M_TOT * D_MODEL);

    gemm(Abf, Win, zx, M_TOT, ZX_LD, D_MODEL);

    conv_silu<<<(M_TOT * CONV_DIM + 255) / 256, 256, 0, stream>>>(zx, conv_w,
                                                                  conv_b, xbc);
    dt_da_kernel<<<(M_TOT * NHEADS) / 256, 256, 0, stream>>>(zx, dt_bias, A_log,
                                                             dt, dA);
    scan_kernel<<<NGRAPHS * NHEADS, HEADDIM, 0, stream>>>(xbc, dt, dA, Dskip,
                                                          ybuf);
    gate_rms<<<M_TOT, 256, 0, stream>>>(ybuf, zx, rms_w, Abf);

    gemm(Abf, Wout, seq, M_TOT, D_MODEL, D_INNER);
  }

  // ---- masked mean pool + layernorm ---------------------------------------
  pool_ln<<<NGRAPHS, D_MODEL, 0, stream>>>(seq, lens, ln_w, ln_b, (float*)d_out);
}
